// WindowClassAttention_48524540510916
// MI455X (gfx1250) — matrix operand order repaired and checked by reading, not compiled, not run on hardware
//
#include <hip/hip_runtime.h>
#include <hip/hip_bf16.h>
#include <cstdint>

// ---------------------------------------------------------------------------
// Problem constants (from reference)
// ---------------------------------------------------------------------------
constexpr int B_   = 512;
constexpr int N_   = 49;
constexpr int C_   = 256;
constexpr int H_   = 8;
constexpr int DH_  = 32;
constexpr int RB_  = 8;
constexpr int NRF_ = 32;
constexpr int NWIN_= 64;
constexpr int DC_  = 64;
constexpr int TC_  = 384;
constexpr int MR_  = B_ * N_;          // 25088 rows
constexpr int NP_  = 64;               // padded window length (49 -> 64)
constexpr float SCALE_ = 0.17677669529663687f;   // 32^-0.5

typedef __attribute__((ext_vector_type(16))) _Float16 v16h;
typedef __attribute__((ext_vector_type(8)))  _Float16 v8h;
typedef __attribute__((ext_vector_type(8)))  float    v8f;

// ---------------------------------------------------------------------------
// Generic batched WMMA GEMM:  D = A(MxK,f16) * Bt(NcxK,f16)^T  (+bias +addC)
// One wave computes a 32x32 block of D as 2x2 WMMA tiles (register blocking:
// each A fragment is reused across 2 N-tiles and each B fragment across 2
// M-tiles -> 4 v_wmma per 8 fragment loads, 2x the arithmetic intensity of
// a single-tile wave).  Batch strides are two-level:
//   base(P) = (z / bdiv) * p1 + (z % bdiv) * p2
// which expresses both linear batching (p1 = bdiv*s, p2 = s) and
// head-interleaved output layouts.  Requires M%32==0 and Nc%32==0.
// ---------------------------------------------------------------------------
__global__ __launch_bounds__(32) void gemm_wmma_f16(
    const _Float16* __restrict__ A,  long a1, long a2, int lda,
    const _Float16* __restrict__ Bt, long b1, long b2, int ldb,
    float* __restrict__ D,           long d1, long d2, int ldd,
    const float* __restrict__ bias,
    const float* __restrict__ addC,  long c1, long c2, int ldc,
    int bdiv, int K, int Mvalid)
{
    const int lane = threadIdx.x & 31;
    const int tn = blockIdx.x;
    const int tm = blockIdx.y;
    const int z  = blockIdx.z;
    const int zq = z / bdiv, zr = z % bdiv;

    const _Float16* Ab = A  + zq * a1 + zr * a2;
    const _Float16* Bb = Bt + zq * b1 + zr * b2;
    float*          Db = D  + zq * d1 + zr * d2;

    const int lh = lane >> 4;      // half-wave: 0 or 1
    const int ll = lane & 15;

    const int rowA0 = tm * 32 + ll;        // A rows for this lane (2 M-tiles)
    const int col0  = tn * 32 + ll;        // D/B columns (2 N-tiles)
    const _Float16* arow0 = Ab + (long)rowA0 * lda;
    const _Float16* arow1 = arow0 + 16L * lda;
    const _Float16* brow0 = Bb + (long)col0 * ldb;
    const _Float16* brow1 = brow0 + 16L * ldb;

    v8f acc00 = {}, acc01 = {}, acc10 = {}, acc11 = {};

    for (int k0 = 0; k0 < K; k0 += 32) {
        // ISA 16-bit A layout: half-wave lh needs K = k0+lh*8 .. +8 and
        // k0+16+lh*8 .. +8 -> two contiguous 16B loads per fragment.
        v8h a0lo = *(const v8h*)(arow0 + k0 + lh * 8);
        v8h a0hi = *(const v8h*)(arow0 + k0 + 16 + lh * 8);
        v8h a1lo = *(const v8h*)(arow1 + k0 + lh * 8);
        v8h a1hi = *(const v8h*)(arow1 + k0 + 16 + lh * 8);
        // B layout: same K order as A: k0+lh*8 .. +8 and k0+16+lh*8 .. +8
        // (Bt is pre-transposed [Nc, K], so each half is contiguous).
        v8h b0lo = *(const v8h*)(brow0 + k0 + lh * 8);
        v8h b0hi = *(const v8h*)(brow0 + k0 + 16 + lh * 8);
        v8h b1lo = *(const v8h*)(brow1 + k0 + lh * 8);
        v8h b1hi = *(const v8h*)(brow1 + k0 + 16 + lh * 8);

        __builtin_prefetch(arow0 + k0 + 64, 0, 3);   // global_prefetch_b8
        __builtin_prefetch(arow1 + k0 + 64, 0, 3);
        __builtin_prefetch(brow0 + k0 + 64, 0, 3);
        __builtin_prefetch(brow1 + k0 + 64, 0, 3);

        v16h a0 = __builtin_shufflevector(a0lo, a0hi, 0,1,2,3,4,5,6,7,8,9,10,11,12,13,14,15);
        v16h a1 = __builtin_shufflevector(a1lo, a1hi, 0,1,2,3,4,5,6,7,8,9,10,11,12,13,14,15);
        v16h bf0 = __builtin_shufflevector(b0lo, b0hi, 0,1,2,3,4,5,6,7,8,9,10,11,12,13,14,15);
        v16h bf1 = __builtin_shufflevector(b1lo, b1hi, 0,1,2,3,4,5,6,7,8,9,10,11,12,13,14,15);

        acc00 = __builtin_amdgcn_wmma_f32_16x16x32_f16(false, a0, false, bf0, (short)0, acc00, false, false);
        acc01 = __builtin_amdgcn_wmma_f32_16x16x32_f16(false, a0, false, bf1, (short)0, acc01, false, false);
        acc10 = __builtin_amdgcn_wmma_f32_16x16x32_f16(false, a1, false, bf0, (short)0, acc10, false, false);
        acc11 = __builtin_amdgcn_wmma_f32_16x16x32_f16(false, a1, false, bf1, (short)0, acc11, false, false);
    }

    const float bb0 = bias ? bias[col0] : 0.0f;
    const float bb1 = bias ? bias[col0 + 16] : 0.0f;
    const float* addb = addC ? (addC + zq * c1 + zr * c2) : nullptr;

    #pragma unroll
    for (int v = 0; v < 8; ++v) {
        const int row0 = tm * 32 + v + lh * 8;  // ISA C/D layout, M-tile 0
        const int row1 = row0 + 16;             // M-tile 1
        if (row0 < Mvalid) {
            float v00 = acc00[v] + bb0;
            float v01 = acc01[v] + bb1;
            if (addb) {
                v00 += addb[(long)row0 * ldc + col0];
                v01 += addb[(long)row0 * ldc + col0 + 16];
            }
            Db[(long)row0 * ldd + col0]      = v00;
            Db[(long)row0 * ldd + col0 + 16] = v01;
        }
        if (row1 < Mvalid) {
            float v10 = acc10[v] + bb0;
            float v11 = acc11[v] + bb1;
            if (addb) {
                v10 += addb[(long)row1 * ldc + col0];
                v11 += addb[(long)row1 * ldc + col0 + 16];
            }
            Db[(long)row1 * ldd + col0]      = v10;
            Db[(long)row1 * ldd + col0 + 16] = v11;
        }
    }
}

// ---------------------------------------------------------------------------
// Elementwise / helper kernels
// ---------------------------------------------------------------------------
__global__ void k_f2h(const float* __restrict__ s, _Float16* __restrict__ d,
                      long n, float scale)
{
    long i = (long)blockIdx.x * blockDim.x + threadIdx.x;
    if (i < n) d[i] = (_Float16)(s[i] * scale);
}

__global__ void k_zeroh(_Float16* __restrict__ d, long n)
{
    long i = (long)blockIdx.x * blockDim.x + threadIdx.x;
    if (i < n) d[i] = (_Float16)0.0f;
}

// weight [K, Nc] f32 -> [Nc, K] f16 (pre-transposed B for the GEMM)
__global__ void k_tconv(const float* __restrict__ w, _Float16* __restrict__ wt,
                        int K, int Nc)
{
    long i = (long)blockIdx.x * blockDim.x + threadIdx.x;
    if (i >= (long)K * Nc) return;
    int k = (int)(i / Nc), n = (int)(i % Nc);
    wt[(long)n * K + k] = (_Float16)w[i];
}

// qkv [MR,768] f32 -> qsA f16 [64][3136][32] (scaled q, ref-branch A),
//                     q_sc f32 [MR,256] (unscaled shortcut),
//                     khpad f16 [4096][64][32] (padded K rows = Bt of scores),
//                     vt f16 [4096][32][64] (V^T padded = Bt of attn@V)
__global__ void k_split_qkv(const float* __restrict__ qkv,
                            _Float16* __restrict__ qsA,
                            float* __restrict__ q_sc,
                            _Float16* __restrict__ khpad,
                            _Float16* __restrict__ vt)
{
    long i = (long)blockIdx.x * blockDim.x + threadIdx.x;
    if (i >= (long)MR_ * C_) return;
    int j   = (int)(i & 255);
    long row = i >> 8;
    int h = j >> 5, d = j & 31;
    int b = (int)(row / N_), n = (int)(row % N_);
    const float* base = qkv + row * 768;
    float q = base[j], k = base[256 + j], v = base[512 + j];
    // (r,h) batch index; w = b % 64
    qsA[(long)((b >> 6) * 8 + h) * (3136L * 32) + ((long)((b & 63) * N_ + n) * 32 + d)]
        = (_Float16)(q * SCALE_);
    q_sc[i] = q;
    long bh = (long)b * 8 + h;
    khpad[bh * 2048 + n * 32 + d] = (_Float16)k;
    vt[bh * 2048 + d * 64 + n]    = (_Float16)v;
}

// ref_qk [256,512] f32 -> refqt f16 [64][32][32] (Bt of ref scores)
//                         refvt f16 [64][32][32] (Bt of ref attn@V)
__global__ void k_reftrans(const float* __restrict__ refqk,
                           const float* __restrict__ mu,
                           const float* __restrict__ ls,
                           _Float16* __restrict__ refqt,
                           _Float16* __restrict__ refvt)
{
    int i = blockIdx.x * blockDim.x + threadIdx.x;
    if (i >= 256 * 256) return;
    int c = i & 255, rowi = i >> 8;         // rowi = r*32+f
    int r = rowi >> 5, f = rowi & 31;
    int h = c >> 5, d = c & 31;
    float rq = mu[c] + __expf(ls[c]) * refqk[(long)rowi * 512 + c];
    float rv = refqk[(long)rowi * 512 + 256 + c];
    refqt[(long)(r * 8 + h) * 1024 + f * 32 + d] = (_Float16)rq;
    refvt[(long)(r * 8 + h) * 1024 + d * 32 + f] = (_Float16)rv;
}

// 3x3 SAME conv, channels = H, over [RB][H][3136][32]
__global__ void k_conv3x3(const float* __restrict__ in,
                          const float* __restrict__ w,
                          const float* __restrict__ bias,
                          float* __restrict__ out)
{
    const long TOT = 64L * 3136 * 32;
    long i = (long)blockIdx.x * blockDim.x + threadIdx.x;
    if (i >= TOT) return;
    int x = (int)(i & 31);
    long t = i >> 5;
    int y  = (int)(t % 3136);
    long t2 = t / 3136;
    int ho = (int)(t2 & 7);
    int r  = (int)(t2 >> 3);
    float acc = bias[ho];
    for (int hi = 0; hi < 8; ++hi) {
        const float* ip = in + ((long)(r * 8 + hi) * 3136) * 32;
        const float* wp = w + (ho * 8 + hi) * 9;
        #pragma unroll
        for (int dy = -1; dy <= 1; ++dy) {
            int yy = y + dy;
            if (yy < 0 || yy >= 3136) continue;
            #pragma unroll
            for (int dx = -1; dx <= 1; ++dx) {
                int xx = x + dx;
                if (xx < 0 || xx >= 32) continue;
                acc += wp[(dy + 1) * 3 + (dx + 1)] * ip[(long)yy * 32 + xx];
            }
        }
    }
    out[i] = acc;
}

// per (r,h) mean/var over 3136*32 = 100352 elements
__global__ __launch_bounds__(256) void k_stats(const float* __restrict__ u,
                                               float* __restrict__ stats)
{
    __shared__ float ss[256], sq[256];
    int sl = blockIdx.x;
    const float* p = u + (long)sl * 100352;
    float s = 0.f, q = 0.f;
    for (int i = threadIdx.x; i < 100352; i += 256) {
        float v = p[i];
        s += v; q += v * v;
    }
    ss[threadIdx.x] = s; sq[threadIdx.x] = q;
    __syncthreads();
    for (int o = 128; o > 0; o >>= 1) {
        if ((int)threadIdx.x < o) {
            ss[threadIdx.x] += ss[threadIdx.x + o];
            sq[threadIdx.x] += sq[threadIdx.x + o];
        }
        __syncthreads();
    }
    if (threadIdx.x == 0) {
        float m = ss[0] / 100352.f;
        stats[sl * 2]     = m;
        stats[sl * 2 + 1] = sq[0] / 100352.f - m * m;
    }
}

__global__ void k_lngelu_add(const float* __restrict__ u,
                             const float* __restrict__ stats,
                             float* __restrict__ attn)
{
    const long TOT = 64L * 3136 * 32;
    long i = (long)blockIdx.x * blockDim.x + threadIdx.x;
    if (i >= TOT) return;
    int sl = (int)(i / 100352);
    float m = stats[sl * 2], var = stats[sl * 2 + 1];
    float x = (u[i] - m) * rsqrtf(var + 1e-5f);
    float g = 0.5f * x * (1.0f + erff(x * 0.70710678118f));
    attn[i] += g;
}

// softmax over rows of 32, f16 out (wave per row)
__global__ void k_smax32(const float* __restrict__ s, _Float16* __restrict__ o,
                         long rows)
{
    long row = (long)blockIdx.x * 8 + threadIdx.y;
    if (row >= rows) return;
    int l = threadIdx.x;
    float v = s[row * 32 + l];
    float m = v;
    for (int off = 16; off; off >>= 1) m = fmaxf(m, __shfl_xor(m, off, 32));
    float e = __expf(v - m);
    float t = e;
    for (int off = 16; off; off >>= 1) t += __shfl_xor(t, off, 32);
    o[row * 32 + l] = (_Float16)(e / t);
}

// softmax over rows of 64, f16 out (wave per row, 2 elems/lane)
__global__ void k_smax64(const float* __restrict__ s, _Float16* __restrict__ o,
                         long rows)
{
    long row = (long)blockIdx.x * 8 + threadIdx.y;
    if (row >= rows) return;
    int l = threadIdx.x;
    float v0 = s[row * 64 + l], v1 = s[row * 64 + 32 + l];
    float m = fmaxf(v0, v1);
    for (int off = 16; off; off >>= 1) m = fmaxf(m, __shfl_xor(m, off, 32));
    float e0 = __expf(v0 - m), e1 = __expf(v1 - m);
    float t = e0 + e1;
    for (int off = 16; off; off >>= 1) t += __shfl_xor(t, off, 32);
    float inv = 1.0f / t;
    o[row * 64 + l]      = (_Float16)(e0 * inv);
    o[row * 64 + 32 + l] = (_Float16)(e1 * inv);
}

// y = LN(qnew + q_sc) * gw + gb, f16 out (wave per row of 256)
__global__ void k_qnew_ln(const float* __restrict__ qn,
                          const float* __restrict__ qsc,
                          const float* __restrict__ gw,
                          const float* __restrict__ gb,
                          _Float16* __restrict__ y)
{
    long row = (long)blockIdx.x * 8 + threadIdx.y;
    if (row >= MR_) return;
    int l = threadIdx.x;
    float t[8];
    float s = 0.f;
    #pragma unroll
    for (int i = 0; i < 8; ++i) {
        int c = i * 32 + l;
        t[i] = qn[row * 256 + c] + qsc[row * 256 + c];
        s += t[i];
    }
    for (int off = 16; off; off >>= 1) s += __shfl_xor(s, off, 32);
    float m = s * (1.0f / 256.0f);
    float q = 0.f;
    #pragma unroll
    for (int i = 0; i < 8; ++i) { float d = t[i] - m; q += d * d; }
    for (int off = 16; off; off >>= 1) q += __shfl_xor(q, off, 32);
    float r = rsqrtf(q * (1.0f / 256.0f) + 1e-5f);
    #pragma unroll
    for (int i = 0; i < 8; ++i) {
        int c = i * 32 + l;
        y[row * 256 + c] = (_Float16)(((t[i] - m) * r) * gw[c] + gb[c]);
    }
}

// qproj_out [MR,256] f32 -> qnh f16 [4096][64][32], scaled + padded
__global__ void k_build_qnh(const float* __restrict__ qp, _Float16* __restrict__ qnh)
{
    long i = (long)blockIdx.x * blockDim.x + threadIdx.x;
    if (i >= 4096L * 64 * 32) return;
    int d = (int)(i & 31);
    long t = i >> 5;
    int np = (int)(t & 63);
    long bh = t >> 6;
    int b = (int)(bh >> 3), h = (int)(bh & 7);
    float val = (np < N_) ? qp[((long)b * N_ + np) * 256 + h * 32 + d] * SCALE_ : 0.0f;
    qnh[i] = (_Float16)val;
}

// relative-position bias, padded 64x64: cols>=49 -> -1e30, rows>=49 -> 0
__global__ void k_bias_build(const float* __restrict__ rb, float* __restrict__ bp)
{
    int i = blockIdx.x * blockDim.x + threadIdx.x;
    if (i >= 8 * 64 * 64) return;
    int j = i & 63;
    int t = i >> 6;
    int ii = t & 63;
    int h = t >> 6;
    float val;
    if (j >= N_)       val = -1e30f;
    else if (ii >= N_) val = 0.0f;
    else {
        int yi = ii / 7, xi = ii % 7, yj = j / 7, xj = j % 7;
        int idx = (yi - yj + 6) * 13 + (xi - xj + 6);
        val = rb[idx * 8 + h];
    }
    bp[i] = val;
}

// concat(out, depth_token, seg_token) -> f16 [MR, 384]
__global__ void k_build_txh(const float* __restrict__ out,
                            const float* __restrict__ dt,
                            const float* __restrict__ st,
                            _Float16* __restrict__ tx)
{
    long i = (long)blockIdx.x * blockDim.x + threadIdx.x;
    if (i >= (long)MR_ * TC_) return;
    int j = (int)(i % TC_);
    long row = i / TC_;
    float v;
    if (j < 256)      v = out[row * 256 + j];
    else if (j < 320) v = dt[row * 64 + (j - 256)];
    else              v = st[row * 64 + (j - 320)];
    tx[i] = (_Float16)v;
}

// class-token attention: per (b,h) block. tiny 8x48 / 8x49 GEMMs -> VALU.
__global__ __launch_bounds__(128) void k_cls_attn(
    const float* __restrict__ dq_, const float* __restrict__ sq_,
    const float* __restrict__ tk_, const float* __restrict__ tv_,
    _Float16* __restrict__ dpre, _Float16* __restrict__ spre)
{
    int bh = blockIdx.x;
    int b = bh >> 3, h = bh & 7;
    __shared__ float tk[49][48], tv[49][48], dq[49][8], sq[49][8];
    __shared__ float aD[8][48], aS[8][48];
    int tid = threadIdx.x;
    for (int i = tid; i < 49 * 48; i += 128) {
        int n = i / 48, t = i % 48;
        long off = ((long)b * N_ + n) * TC_ + h * 48 + t;
        tk[n][t] = tk_[off];
        tv[n][t] = tv_[off];
    }
    for (int i = tid; i < 49 * 8; i += 128) {
        int n = i / 8, c = i % 8;
        long off = ((long)b * N_ + n) * 64 + h * 8 + c;
        dq[n][c] = dq_[off] * SCALE_;
        sq[n][c] = sq_[off] * SCALE_;
    }
    __syncthreads();
    for (int i = tid; i < 8 * 48; i += 128) {
        int c = i / 48, t = i % 48;
        float s1 = 0.f, s2 = 0.f;
        for (int n = 0; n < 49; ++n) {
            s1 += dq[n][c] * tk[n][t];
            s2 += sq[n][c] * tk[n][t];
        }
        aD[c][t] = s1; aS[c][t] = s2;
    }
    __syncthreads();
    if (tid < 16) {
        float* rowp = (tid < 8) ? aD[tid] : aS[tid - 8];
        float m = -1e30f;
        for (int t = 0; t < 48; ++t) m = fmaxf(m, rowp[t]);
        float s = 0.f;
        for (int t = 0; t < 48; ++t) { rowp[t] = __expf(rowp[t] - m); s += rowp[t]; }
        float inv = 1.0f / s;
        for (int t = 0; t < 48; ++t) rowp[t] *= inv;
    }
    __syncthreads();
    for (int i = tid; i < 8 * 49; i += 128) {
        int c = i / 49, n = i % 49;
        float s1 = 0.f, s2 = 0.f;
        for (int t = 0; t < 48; ++t) {
            s1 += aD[c][t] * tv[n][t];
            s2 += aS[c][t] * tv[n][t];
        }
        long off = ((long)b * N_ + n) * 64 + h * 8 + c;
        dpre[off] = (_Float16)s1;
        spre[off] = (_Float16)s2;
    }
}

// ---------------------------------------------------------------------------
// Host side
// ---------------------------------------------------------------------------
static inline void launch_gemm(hipStream_t st,
    const _Float16* A, long a1, long a2, int lda,
    const _Float16* Bt, long b1, long b2, int ldb,
    float* D, long d1, long d2, int ldd,
    const float* bias,
    const float* addC, long c1, long c2, int ldc,
    int bdiv, int M, int Nc, int K, int Mvalid, int batch)
{
    dim3 g((unsigned)(Nc / 32), (unsigned)(M / 32), (unsigned)batch);
    gemm_wmma_f16<<<g, 32, 0, st>>>(A, a1, a2, lda, Bt, b1, b2, ldb,
                                    D, d1, d2, ldd, bias,
                                    addC, c1, c2, ldc, bdiv, K, Mvalid);
}

static inline dim3 ew_grid(long n) { return dim3((unsigned)((n + 255) / 256)); }

extern "C" void kernel_launch(void* const* d_in, const int* in_sizes, int n_in,
                              void* d_out, int out_size, void* d_ws, size_t ws_size,
                              hipStream_t stream)
{
    (void)in_sizes; (void)n_in; (void)out_size; (void)ws_size;
    // ---- inputs (all f32, dict order) ----
    const float* x        = (const float*)d_in[0];
    const float* x_ref    = (const float*)d_in[1];
    const float* dtok     = (const float*)d_in[2];
    const float* stok     = (const float*)d_in[3];
    const float* qkv_w    = (const float*)d_in[4];
    const float* qkv_b    = (const float*)d_in[5];
    const float* proj_w   = (const float*)d_in[6];
    const float* proj_b   = (const float*)d_in[7];
    const float* rel_bias = (const float*)d_in[8];
    const float* diff_mu  = (const float*)d_in[9];
    const float* diff_ls  = (const float*)d_in[10];
    const float* refqk_w  = (const float*)d_in[11];
    const float* refqk_b  = (const float*)d_in[12];
    const float* conv_w   = (const float*)d_in[13];
    const float* conv_b   = (const float*)d_in[14];
    const float* qn_w     = (const float*)d_in[15];
    const float* qn_b     = (const float*)d_in[16];
    const float* qproj_w  = (const float*)d_in[17];
    const float* qproj_b  = (const float*)d_in[18];
    const float* dq_w     = (const float*)d_in[19];
    const float* dq_b     = (const float*)d_in[20];
    const float* sq_w     = (const float*)d_in[21];
    const float* sq_b     = (const float*)d_in[22];
    const float* gk_w     = (const float*)d_in[23];
    const float* gk_b     = (const float*)d_in[24];
    const float* gv_w     = (const float*)d_in[25];
    const float* gv_b     = (const float*)d_in[26];
    const float* pd_w     = (const float*)d_in[27];
    const float* pd_b     = (const float*)d_in[28];

    float* out_final = (float*)d_out;                       // [MR, 256]
    float* dtok_out  = out_final + (long)MR_ * C_;          // [MR, 64]
    float* stok_out  = dtok_out + (long)MR_ * DC_;          // [MR, 64]

    // ---- workspace bump allocator ----
    char* wsb = (char*)d_ws;
    size_t cur = 0;
    auto alloc = [&](size_t bytes) -> void* {
        void* p = wsb + cur;
        cur = (cur + bytes + 255) & ~(size_t)255;
        return p;
    };
    auto ah = [&](long elems) { return (_Float16*)alloc((size_t)elems * 2); };
    auto af = [&](long elems) { return (float*)alloc((size_t)elems * 4); };

    // f16 staging
    _Float16* xh      = ah((long)MR_ * C_);
    _Float16* qkvw_t  = ah(768L * 256);
    _Float16* refqkw_t= ah(512L * 256);
    _Float16* xrefh   = ah(256L * 256);
    _Float16* qprojw_t= ah(256L * 256);
    _Float16* projw_t = ah(256L * 256);
    _Float16* gkw_t   = ah(384L * 384);
    _Float16* gvw_t   = ah(384L * 384);
    _Float16* dqw_t   = ah(64L * 64);
    _Float16* sqw_t   = ah(64L * 64);
    _Float16* pdw_t   = ah(64L * 64);
    _Float16* dth_h   = ah((long)MR_ * DC_);
    _Float16* sth_h   = ah((long)MR_ * DC_);
    _Float16* qsA     = ah(64L * 3136 * 32);
    _Float16* khpad   = ah(4096L * 64 * 32);
    _Float16* vt      = ah(4096L * 32 * 64);
    _Float16* refqt   = ah(64L * 1024);
    _Float16* refvt   = ah(64L * 1024);
    _Float16* refat_h = ah(64L * 3136 * 32);
    _Float16* ynh     = ah((long)MR_ * 256);
    _Float16* qnh     = ah(4096L * 64 * 32);
    _Float16* attn_h  = ah(4096L * 64 * 64);
    _Float16* outpreh = ah((long)MR_ * 256);
    _Float16* txh     = ah((long)MR_ * TC_);
    _Float16* dpreh   = ah((long)MR_ * DC_);
    _Float16* spreh   = ah((long)MR_ * DC_);
    // f32 intermediates
    float* qkv_f    = af((long)MR_ * 768);
    float* q_sc     = af((long)MR_ * 256);
    float* refqk_f  = af(256L * 512);
    float* refat_f  = af(64L * 3136 * 32);
    float* conv_u   = af(64L * 3136 * 32);
    float* stats    = af(128);
    float* qnew_f   = af((long)MR_ * 256);
    float* qproj_f  = af((long)MR_ * 256);
    float* winsc_f  = af(4096L * 64 * 64);
    float* outpre_f = af((long)MR_ * 256);
    float* biaspad  = af(8L * 64 * 64);
    float* tk_f     = af((long)MR_ * TC_);
    float* tv_f     = af((long)MR_ * TC_);
    float* dq_f     = af((long)MR_ * DC_);
    float* sq_f     = af((long)MR_ * DC_);

    // ---- stage 0: weight transposes & input conversions ----
    k_tconv<<<ew_grid(256L*768), 256, 0, stream>>>(qkv_w,   qkvw_t,   256, 768);
    k_tconv<<<ew_grid(256L*512), 256, 0, stream>>>(refqk_w, refqkw_t, 256, 512);
    k_tconv<<<ew_grid(256L*256), 256, 0, stream>>>(qproj_w, qprojw_t, 256, 256);
    k_tconv<<<ew_grid(256L*256), 256, 0, stream>>>(proj_w,  projw_t,  256, 256);
    k_tconv<<<ew_grid(384L*384), 256, 0, stream>>>(gk_w,    gkw_t,    384, 384);
    k_tconv<<<ew_grid(384L*384), 256, 0, stream>>>(gv_w,    gvw_t,    384, 384);
    k_tconv<<<ew_grid(64L*64),   256, 0, stream>>>(dq_w,    dqw_t,    64,  64);
    k_tconv<<<ew_grid(64L*64),   256, 0, stream>>>(sq_w,    sqw_t,    64,  64);
    k_tconv<<<ew_grid(64L*64),   256, 0, stream>>>(pd_w,    pdw_t,    64,  64);
    k_f2h<<<ew_grid((long)MR_*C_), 256, 0, stream>>>(x,     xh,    (long)MR_*C_, 1.0f);
    k_f2h<<<ew_grid(256L*256),     256, 0, stream>>>(x_ref, xrefh, 256L*256,     1.0f);
    k_f2h<<<ew_grid((long)MR_*DC_),256, 0, stream>>>(dtok,  dth_h, (long)MR_*DC_,1.0f);
    k_f2h<<<ew_grid((long)MR_*DC_),256, 0, stream>>>(stok,  sth_h, (long)MR_*DC_,1.0f);
    k_bias_build<<<ew_grid(8L*64*64), 256, 0, stream>>>(rel_bias, biaspad);
    k_zeroh<<<ew_grid(4096L*2048), 256, 0, stream>>>(khpad, 4096L*2048);
    k_zeroh<<<ew_grid(4096L*2048), 256, 0, stream>>>(vt,    4096L*2048);

    // ---- stage 1: qkv projection (WMMA) + split ----
    launch_gemm(stream, xh, 0,0, 256, qkvw_t, 0,0, 256,
                qkv_f, 0,0, 768, qkv_b, nullptr,0,0,0,
                1, MR_, 768, 256, MR_, 1);
    k_split_qkv<<<ew_grid((long)MR_*C_), 256, 0, stream>>>(qkv_f, qsA, q_sc, khpad, vt);

    // ---- stage 2: ref branch projection (WMMA) + transform ----
    launch_gemm(stream, xrefh, 0,0, 256, refqkw_t, 0,0, 256,
                refqk_f, 0,0, 512, refqk_b, nullptr,0,0,0,
                1, 256, 512, 256, 256, 1);
    k_reftrans<<<ew_grid(256L*256), 256, 0, stream>>>(refqk_f, diff_mu, diff_ls,
                                                      refqt, refvt);

    // ---- stage 3: ref scores, 64 batched WMMA GEMMs [3136x32]x[32x32] ----
    launch_gemm(stream, qsA, 3136L*32, 0, 32, refqt, 1024, 0, 32,
                refat_f, 3136L*32, 0, 32, nullptr, nullptr,0,0,0,
                1, 3136, 32, 32, 3136, 64);

    // ---- stage 4: 3x (conv3x3 -> per-slice LN -> gelu -> residual) ----
    for (int it = 0; it < 3; ++it) {
        k_conv3x3<<<ew_grid(64L*3136*32), 256, 0, stream>>>(refat_f, conv_w, conv_b, conv_u);
        k_stats<<<64, 256, 0, stream>>>(conv_u, stats);
        k_lngelu_add<<<ew_grid(64L*3136*32), 256, 0, stream>>>(conv_u, stats, refat_f);
    }

    // ---- stage 5: softmax(32) -> f16 ----
    {
        long rows = 64L * 3136;
        dim3 blk(32, 8);
        k_smax32<<<dim3((unsigned)((rows + 7) / 8)), blk, 0, stream>>>(refat_f, refat_h, rows);
    }

    // ---- stage 6: q_new = attn @ ref_v, scatter to [B,N,C] via 2-level stride ----
    launch_gemm(stream, refat_h, 8L*3136*32, 3136L*32, 32,
                refvt, 8L*1024, 1024, 32,
                qnew_f, 3136L*256, 32, 256, nullptr, nullptr,0,0,0,
                8, 3136, 32, 32, 3136, 64);

    // ---- stage 7: LN + q_proj (WMMA, residual = q_sc) ----
    {
        dim3 blk(32, 8);
        k_qnew_ln<<<dim3((unsigned)((MR_ + 7) / 8)), blk, 0, stream>>>(qnew_f, q_sc, qn_w, qn_b, ynh);
    }
    launch_gemm(stream, ynh, 0,0, 256, qprojw_t, 0,0, 256,
                qproj_f, 0,0, 256, qproj_b, q_sc, 0,0, 256,
                1, MR_, 256, 256, MR_, 1);
    k_build_qnh<<<ew_grid(4096L*64*32), 256, 0, stream>>>(qproj_f, qnh);

    // ---- stage 8: window attention, 4096 batched WMMA GEMMs (padded 49->64) ----
    launch_gemm(stream, qnh, 8L*2048, 2048, 32, khpad, 8L*2048, 2048, 32,
                winsc_f, 8L*4096, 4096, 64, nullptr,
                biaspad, 0, 64L*64, 64,
                8, 64, 64, 32, 64, 4096);
    {
        long rows = 4096L * 64;
        dim3 blk(32, 8);
        k_smax64<<<dim3((unsigned)((rows + 7) / 8)), blk, 0, stream>>>(winsc_f, attn_h, rows);
    }
    launch_gemm(stream, attn_h, 8L*4096, 4096, 64, vt, 8L*2048, 2048, 64,
                outpre_f, (long)N_*256, 32, 256, nullptr, nullptr,0,0,0,
                8, 64, 32, 64, N_ /*mask padded rows*/, 4096);

    // ---- stage 9: output projection (WMMA) -> d_out[0:MR*C] ----
    k_f2h<<<ew_grid((long)MR_*256), 256, 0, stream>>>(outpre_f, outpreh, (long)MR_*256, 1.0f);
    launch_gemm(stream, outpreh, 0,0, 256, projw_t, 0,0, 256,
                out_final, 0,0, 256, proj_b, nullptr,0,0,0,
                1, MR_, 256, 256, MR_, 1);

    // ---- stage 10: class-token global attention ----
    k_build_txh<<<ew_grid((long)MR_*TC_), 256, 0, stream>>>(out_final, dtok, stok, txh);
    launch_gemm(stream, txh, 0,0, 384, gkw_t, 0,0, 384,
                tk_f, 0,0, 384, gk_b, nullptr,0,0,0,
                1, MR_, 384, 384, MR_, 1);
    launch_gemm(stream, txh, 0,0, 384, gvw_t, 0,0, 384,
                tv_f, 0,0, 384, gv_b, nullptr,0,0,0,
                1, MR_, 384, 384, MR_, 1);
    launch_gemm(stream, dth_h, 0,0, 64, dqw_t, 0,0, 64,
                dq_f, 0,0, 64, dq_b, nullptr,0,0,0,
                1, MR_, 64, 64, MR_, 1);
    launch_gemm(stream, sth_h, 0,0, 64, sqw_t, 0,0, 64,
                sq_f, 0,0, 64, sq_b, nullptr,0,0,0,
                1, MR_, 64, 64, MR_, 1);
    k_cls_attn<<<4096, 128, 0, stream>>>(dq_f, sq_f, tk_f, tv_f, dpreh, spreh);
    launch_gemm(stream, dpreh, 0,0, 64, pdw_t, 0,0, 64,
                dtok_out, 0,0, 64, pd_b, nullptr,0,0,0,
                1, MR_, 64, 64, MR_, 1);
    launch_gemm(stream, spreh, 0,0, 64, pdw_t, 0,0, 64,
                stok_out, 0,0, 64, pd_b, nullptr,0,0,0,
                1, MR_, 64, 64, MR_, 1);
}
